// AUwGCN_31207232373169
// MI455X (gfx1250) — matrix operand order repaired and checked by reading, not compiled, not run on hardware
//
#include <hip/hip_runtime.h>

typedef __attribute__((ext_vector_type(16))) _Float16 v16h;
typedef __attribute__((ext_vector_type(8)))  _Float16 v8h;
typedef __attribute__((ext_vector_type(8)))  float    v8f;
typedef __attribute__((ext_vector_type(4)))  unsigned int u32x4;
typedef __attribute__((ext_vector_type(8)))  int      i32x8;
typedef __attribute__((ext_vector_type(4)))  int      i32x4;

#define B_    32
#define T_    1024
#define NB    (B_ * T_)      // 32768 instances
#define N_    12
#define HID   32
#define HEADS 4
#define SLOPE 0.01f
#define EPS   1e-5f
#define CNT1  ((float)(NB * N_))
#define CNT2  ((float)NB)

// ---------------- WMMA helpers (wave32, 16x16x32 f16 -> f32) ----------------

__device__ inline v8f wmma_f16(v16h a, v16h b, v8f c) {
  return __builtin_amdgcn_wmma_f32_16x16x32_f16(false, a, false, b, (short)0, c,
                                                false, false);
}

// A fragment from row-major f16 [16][ld]. Lane L<16: row M=L, K in
// {kh..kh+7, 16+kh..16+kh+7} with kh = (L>=16)*8 -> two contiguous v8h loads.
__device__ inline v16h load_aT(const _Float16* s, int ld, int lane) {
  int m  = lane & 15;
  int kh = (lane >> 4) * 8;
  v8h lo = *(const v8h*)(s + m * ld + kh);
  v8h hi = *(const v8h*)(s + m * ld + 16 + kh);
  return __builtin_shufflevector(lo, hi, 0, 1, 2, 3, 4, 5, 6, 7,
                                 8, 9, 10, 11, 12, 13, 14, 15);
}

// B fragment from TRANSPOSED f16 source sT[n][k] with leading dim ld.
// Lane: N = lane&15, K in {kh..kh+7, 16+kh..16+kh+7}, kh = (lane>>4)*8 -> two v8h loads.
__device__ inline v16h load_bT(const _Float16* sT, int ld, int lane) {
  int n  = lane & 15;
  int kh = (lane >> 4) * 8;
  v8h lo = *(const v8h*)(sT + n * ld + kh);
  v8h hi = *(const v8h*)(sT + n * ld + 16 + kh);
  return __builtin_shufflevector(lo, hi, 0, 1, 2, 3, 4, 5, 6, 7,
                                 8, 9, 10, 11, 12, 13, 14, 15);
}

// Pack a C fragment column (8 rows, one n) into v8h.
__device__ inline v8h pack8(v8f c) {
  v8h p;
#pragma unroll
  for (int r = 0; r < 8; ++r) p[r] = (_Float16)c[r];
  return p;
}

// ---------------- Kernel 1: GCN layer 0 (pre-BN) + BN1 stats ----------------

__global__ __launch_bounds__(128) void k_gcn0(
    const float* __restrict__ x, const float* __restrict__ adj,
    const float* __restrict__ W0, const float* __restrict__ b0,
    _Float16* __restrict__ ypre,  // padded [inst][16][32]
    float* __restrict__ bn1sum, float* __restrict__ bn1sq) {
  __shared__ alignas(32) _Float16 s_adjA[16][32];   // A-tile (row-major)
  __shared__ alignas(32) _Float16 s_BT[4][32][32];  // xW0 transposed [f][m]
  __shared__ float s_W0[2][32];
  __shared__ float s_b0[32];
  __shared__ float s_sum[32], s_sq[32];

  int tid = threadIdx.x, lane = tid & 31, w = tid >> 5;

  for (int i = tid; i < 512; i += 128) {
    int m = i >> 5, k = i & 31;
    s_adjA[m][k] = (_Float16)((m < N_ && k < N_) ? adj[m * N_ + k] : 0.0f);
  }
  if (tid < 64) s_W0[tid >> 5][tid & 31] = W0[tid];
  if (tid < 32) { s_b0[tid] = b0[tid]; s_sum[tid] = 0.f; s_sq[tid] = 0.f; }
  __syncthreads();

  int inst = blockIdx.x * 4 + w;
  const float* xi = x + inst * (N_ * 2);
  {
    int f = lane;  // each lane owns one feature column -> contiguous row in s_BT
#pragma unroll
    for (int m = 0; m < N_; ++m)
      s_BT[w][f][m] =
          (_Float16)(xi[m * 2] * s_W0[0][f] + xi[m * 2 + 1] * s_W0[1][f]);
#pragma unroll
    for (int m = N_; m < 32; ++m) s_BT[w][f][m] = (_Float16)0.f;
  }
  __syncthreads();

  v16h a = load_aT(&s_adjA[0][0], 32, lane);
  int n = lane & 15, mb = (lane >> 4) * 8;
#pragma unroll
  for (int j = 0; j < 2; ++j) {
    v16h bf = load_bT(&s_BT[w][16 * j][0], 32, lane);
    v8f c = {};
    c = wmma_f16(a, bf, c);
    int f = 16 * j + n;
    float ls = 0.f, ls2 = 0.f;
#pragma unroll
    for (int r = 0; r < 8; ++r) {
      int m = mb + r;
      float v = c[r] + s_b0[f];
      ypre[inst * (16 * HID) + m * HID + f] = (_Float16)v;  // branchless store
      float vm = (m < N_) ? v : 0.f;                        // arithmetic mask
      ls += vm; ls2 += vm * vm;
    }
    atomicAdd(&s_sum[f], ls);
    atomicAdd(&s_sq[f], ls2);
  }
  __syncthreads();
  if (tid < 32) {
    atomicAdd(&bn1sum[tid], s_sum[tid]);
    atomicAdd(&bn1sq[tid], s_sq[tid]);
  }
}

// ------- Kernel 2: BN1+ReLU, GCN1+residual, 4-head GAT, hcat@Wmh -> hout ----

__global__ __launch_bounds__(64) void k_main(
    const float* __restrict__ x, const float* __restrict__ adj,
    const float* __restrict__ W1, const float* __restrict__ b1,
    const float* __restrict__ Wr, const float* __restrict__ br,
    const float* __restrict__ g_bn, const float* __restrict__ be_bn,
    const float* __restrict__ Watt, const float* __restrict__ a_l,
    const float* __restrict__ a_r, const float* __restrict__ Wmh,
    const _Float16* __restrict__ ypre, const float* __restrict__ bn1sum,
    const float* __restrict__ bn1sq, _Float16* __restrict__ hout) {
  // Weights, transposed for B-fragment vector loads: [n][k]
  __shared__ alignas(32) _Float16 s_W1T[32][32];
  __shared__ alignas(32) _Float16 s_WattT[4][32][32];
  __shared__ alignas(32) _Float16 s_WmhT[32][160];
  __shared__ alignas(32) _Float16 s_adjA[16][32];
  // Per-wave tiles. A-role: row-major [m][k]; B-role: transposed [n][k].
  __shared__ alignas(32) _Float16 s_y[2][16][32];
  __shared__ alignas(32) _Float16 s_uT[2][32][32];
  __shared__ alignas(32) _Float16 s_h[2][16][32];
  __shared__ alignas(32) _Float16 s_WhT[2][32][32];
  __shared__ alignas(32) _Float16 s_att[2][16][32];
  __shared__ alignas(32) _Float16 s_hp[2][16][32];
  __shared__ float s_al[4][32], s_ar[4][32];
  __shared__ float s_b1[32], s_br[32], s_Wr[2][32];
  __shared__ float s_scale[32], s_shift[32];
  __shared__ float s_elr[2][32];

  int tid = threadIdx.x, lane = tid & 31, w = tid >> 5;

  for (int i = tid; i < 1024; i += 64) {
    int n = i >> 5, k = i & 31;
    s_W1T[n][k] = (_Float16)W1[k * 32 + n];
  }
  for (int i = tid; i < 4096; i += 64) {
    int h = i >> 10, o = (i >> 5) & 31, f = i & 31;
    s_WattT[h][o][f] = (_Float16)Watt[h * 1024 + f * 32 + o];
  }
  for (int i = tid; i < 5120; i += 64) {
    int o = i / 160, k = i % 160;
    s_WmhT[o][k] = (_Float16)Wmh[k * 32 + o];
  }
  for (int i = tid; i < 512; i += 64) {
    int m = i >> 5, k = i & 31;
    s_adjA[m][k] = (_Float16)((m < N_ && k < N_) ? adj[m * N_ + k] : 0.f);
  }
  for (int i = tid; i < 128; i += 64) {
    s_al[i >> 5][i & 31] = a_l[i];
    s_ar[i >> 5][i & 31] = a_r[i];
  }
  if (tid < 32) {
    s_b1[tid] = b1[tid]; s_br[tid] = br[tid];
    s_Wr[0][tid] = Wr[tid]; s_Wr[1][tid] = Wr[32 + tid];
    float mean = bn1sum[tid] / CNT1;
    float var  = bn1sq[tid] / CNT1 - mean * mean;
    float sc   = g_bn[tid] * rsqrtf(var + EPS);
    s_scale[tid] = sc;
    s_shift[tid] = be_bn[tid] - mean * sc;
  }
  // zero pad K rows 16..31 of B-role tiles (contiguous per lane)
#pragma unroll
  for (int k = 16; k < 32; ++k) {
    s_uT[w][lane][k] = (_Float16)0.f;
    s_WhT[w][lane][k] = (_Float16)0.f;
  }
  __syncthreads();

  int inst = blockIdx.x * 2 + w;
  const _Float16* yp = ypre + inst * (16 * HID);
  // y = relu(bn(ypre)), padded 16x32 f16
#pragma unroll
  for (int m = 0; m < 16; ++m) {
    float v = 0.f;
    if (m < N_) {
      v = (float)yp[m * HID + lane] * s_scale[lane] + s_shift[lane];
      v = v > 0.f ? v : 0.f;
    }
    s_y[w][m][lane] = (_Float16)v;
  }
  __syncthreads();
  int n = lane & 15, mb = (lane >> 4) * 8;

  // u = y @ W1  -> transposed tile (v8h store per lane)
  {
    v16h a = load_aT(&s_y[w][0][0], 32, lane);
#pragma unroll
    for (int j = 0; j < 2; ++j) {
      v16h bf = load_bT(&s_W1T[16 * j][0], 32, lane);
      v8f c = {};
      c = wmma_f16(a, bf, c);
      *(v8h*)&s_uT[w][16 * j + n][mb] = pack8(c);
    }
  }
  __syncthreads();

  // h = adj @ u + b1 + residual(x @ Wr + br)
  const float* xi = x + inst * (N_ * 2);
  {
    v16h a = load_aT(&s_adjA[0][0], 32, lane);
#pragma unroll
    for (int j = 0; j < 2; ++j) {
      v16h bf = load_bT(&s_uT[w][16 * j][0], 32, lane);
      v8f c = {};
      c = wmma_f16(a, bf, c);
      int f = 16 * j + n;
#pragma unroll
      for (int r = 0; r < 8; ++r) {
        int m = mb + r;
        float v = c[r] + s_b1[f] + xi[m * 2] * s_Wr[0][f] +
                  xi[m * 2 + 1] * s_Wr[1][f] + s_br[f];
        s_h[w][m][f] = (_Float16)((m < N_) ? v : 0.f);
      }
    }
  }
  __syncthreads();

  v16h ah = load_aT(&s_h[w][0][0], 32, lane);
  v8f acc0 = {}, acc1 = {};
  {  // hout chunk 0: h @ Wmh[0:32]
    acc0 = wmma_f16(ah, load_bT(&s_WmhT[0][0], 160, lane), acc0);
    acc1 = wmma_f16(ah, load_bT(&s_WmhT[16][0], 160, lane), acc1);
  }

  for (int head = 0; head < HEADS; ++head) {
    // Wh = h @ Watt[head]  -> transposed tile
#pragma unroll
    for (int j = 0; j < 2; ++j) {
      v16h bf = load_bT(&s_WattT[head][16 * j][0], 32, lane);
      v8f c = {};
      c = wmma_f16(ah, bf, c);
      *(v8h*)&s_WhT[w][16 * j + n][mb] = pack8(c);
    }
    __syncthreads();
    // el/er:  el[m] = sum_o Wh[m][o] * a_l[o],  Wh[m][o] = WhT[o][m]
    if (lane < N_) {
      float el = 0.f, er = 0.f;
#pragma unroll
      for (int o = 0; o < 32; ++o) {
        float v = (float)s_WhT[w][o][lane];
        el += v * s_al[head][o];
        er += v * s_ar[head][o];
      }
      s_elr[w][lane] = el;
      s_elr[w][16 + lane] = er;
    }
    __syncthreads();
    // softmax(leaky_relu(el_i + er_j)) -> att A-tile (row per lane, contiguous)
    if (lane < 16) {
      int m = lane;
      if (m < N_) {
        float el = s_elr[w][m];
        float e[N_];
        float mx = -1e30f;
#pragma unroll
        for (int jj = 0; jj < N_; ++jj) {
          float t = el + s_elr[w][16 + jj];
          t = t > 0.f ? t : SLOPE * t;
          e[jj] = t;
          mx = t > mx ? t : mx;
        }
        float ssum = 0.f;
#pragma unroll
        for (int jj = 0; jj < N_; ++jj) { e[jj] = __expf(e[jj] - mx); ssum += e[jj]; }
        float inv = 1.f / ssum;
#pragma unroll
        for (int jj = 0; jj < N_; ++jj) s_att[w][m][jj] = (_Float16)(e[jj] * inv);
#pragma unroll
        for (int jj = N_; jj < 32; ++jj) s_att[w][m][jj] = (_Float16)0.f;
      } else {
#pragma unroll
        for (int jj = 0; jj < 32; ++jj) s_att[w][m][jj] = (_Float16)0.f;
      }
    }
    __syncthreads();
    // hp = att @ Wh   (Wh consumed as transposed B tile)
    {
      v16h aatt = load_aT(&s_att[w][0][0], 32, lane);
#pragma unroll
      for (int j = 0; j < 2; ++j) {
        v16h bf = load_bT(&s_WhT[w][16 * j][0], 32, lane);
        v8f c = {};
        c = wmma_f16(aatt, bf, c);
        int f = 16 * j + n;
#pragma unroll
        for (int r = 0; r < 8; ++r) s_hp[w][mb + r][f] = (_Float16)c[r];
      }
    }
    __syncthreads();
    // accumulate hp @ Wmh[(1+head)*32 : +32]
    {
      v16h ahp = load_aT(&s_hp[w][0][0], 32, lane);
      acc0 = wmma_f16(ahp, load_bT(&s_WmhT[0][(1 + head) * 32], 160, lane), acc0);
      acc1 = wmma_f16(ahp, load_bT(&s_WmhT[16][(1 + head) * 32], 160, lane), acc1);
    }
    __syncthreads();
  }

  _Float16* ho = hout + inst * (N_ * HID);
#pragma unroll
  for (int r = 0; r < 8; ++r) {
    int m = mb + r;
    if (m < N_) {
      ho[m * HID + n]      = (_Float16)acc0[r];
      ho[m * HID + 16 + n] = (_Float16)acc1[r];
    }
  }
}

// ------- Kernel 3: [16x384] @ [384x128] per wave (conv1x1) + BN2 stats ------
// A tile (64 rows x 384 f16, contiguous 48KB) is DMA'd into LDS by the Tensor
// Data Mover; waves consume A-fragments from LDS via ds_load_b128.

__global__ __launch_bounds__(128) void k_conv(
    const _Float16* __restrict__ hout, const float* __restrict__ Wc,
    float* __restrict__ z, float* __restrict__ bn2sum,
    float* __restrict__ bn2sq) {
  __shared__ alignas(64) _Float16 s_A[64][384];       // 48 KB A tile
  __shared__ alignas(32) _Float16 s_chunkT[128][32];  // B transposed [ch][k]
  __shared__ float s_sum[128], s_sq[128];
  int tid = threadIdx.x, lane = tid & 31, w = tid >> 5;
  s_sum[tid] = 0.f; s_sq[tid] = 0.f;

  int rowbase = blockIdx.x * 64;

#if __has_builtin(__builtin_amdgcn_tensor_load_to_lds)
  if (w == 0) {
    // D# for a contiguous 1-D tile: 24576 elements of 2 bytes.
    unsigned lds_addr = (unsigned)(size_t)(&s_A[0][0]);  // addr[31:0] = LDS offset
    unsigned long long ga =
        (unsigned long long)(size_t)(hout + (size_t)rowbase * 384);
    u32x4 g0;
    g0[0] = 1u;                                           // count=1 (valid D#)
    g0[1] = lds_addr;                                     // [63:32] lds_addr
    g0[2] = (unsigned)(ga & 0xffffffffu);                 // [95:64] gaddr lo
    g0[3] = ((unsigned)(ga >> 32) & 0x01ffffffu) | (2u << 30);  // gaddr hi | type=2
    i32x8 g1;
    g1[0] = (1 << 16);                       // data_size = 1 (2 bytes)
    g1[1] = (int)((24576u & 0xffffu) << 16); // tensor_dim0[15:0] -> bits[63:48]
    g1[2] = (int)(((24576u >> 16) & 0xffffu) | (1u << 16));  // dim0 hi | tensor_dim1=1
    g1[3] = (int)((24576u & 0xffffu) << 16); // tile_dim0 = 24576 -> bits[127:112]
    g1[4] = 1;                               // tile_dim1 = 1, tile_dim2 = 0
    g1[5] = 24576;                           // tensor_dim0_stride lo32
    g1[6] = 0; g1[7] = 0;
    i32x4 gz = {0, 0, 0, 0};
#if __clang_major__ >= 23
    i32x8 gz8 = {0, 0, 0, 0, 0, 0, 0, 0};
    __builtin_amdgcn_tensor_load_to_lds(g0, g1, gz, gz, gz8, 0);
#else
    __builtin_amdgcn_tensor_load_to_lds(g0, g1, gz, gz, 0);
#endif
    __builtin_amdgcn_s_wait_tensorcnt((short)0);
  }
#else
  {  // fallback: cooperative global -> LDS copy of the A tile
    const v8h* src = (const v8h*)(hout + (size_t)rowbase * 384);
    v8h* dst = (v8h*)&s_A[0][0];
    for (int i = tid; i < (64 * 384) / 8; i += 128) dst[i] = src[i];
  }
#endif
  __syncthreads();  // A tile visible to all waves (and s_sum init done)

  v8f zero = {};
  v8f acc[8];
#pragma unroll
  for (int j = 0; j < 8; ++j) acc[j] = zero;

  for (int kk = 0; kk < 12; ++kk) {
    {  // stage Wc K-chunk transposed: s_chunkT[ch][k] = Wc[ch][kk*32+k]
      int ch = tid;  // 128 threads -> one channel row each, contiguous k
#pragma unroll
      for (int k = 0; k < 32; ++k)
        s_chunkT[ch][k] = (_Float16)Wc[ch * 384 + kk * 32 + k];
    }
    if (kk < 11)  // prefetch next Wc K-chunk (global_prefetch_b8 path)
      __builtin_prefetch(Wc + tid * 384 + (kk + 1) * 32, 0, 0);
    __syncthreads();
    v16h a = load_aT(&s_A[w * 16][0] + kk * 32, 384, lane);
#pragma unroll
    for (int j = 0; j < 8; ++j) {
      v16h bf = load_bT(&s_chunkT[16 * j][0], 32, lane);
      acc[j] = wmma_f16(a, bf, acc[j]);
    }
    __syncthreads();  // before next chunk overwrite
  }

  int n = lane & 15, mb = (lane >> 4) * 8;
#pragma unroll
  for (int j = 0; j < 8; ++j) {
    int ch = 16 * j + n;
    float ls = 0.f, ls2 = 0.f;
#pragma unroll
    for (int r = 0; r < 8; ++r) {
      float v = acc[j][r];
      z[(rowbase + w * 16 + mb + r) * 128 + ch] = v;
      ls += v; ls2 += v * v;
    }
    atomicAdd(&s_sum[ch], ls);
    atomicAdd(&s_sq[ch], ls2);
  }
  __syncthreads();
  atomicAdd(&bn2sum[tid], s_sum[tid]);
  atomicAdd(&bn2sq[tid], s_sq[tid]);
}

// ------- Kernel 4: BN2 + ReLU + temporal mean + final linear ----------------

__global__ __launch_bounds__(128) void k_head(
    const float* __restrict__ z, const float* __restrict__ bn2sum,
    const float* __restrict__ bn2sq, const float* __restrict__ g2,
    const float* __restrict__ be2, const float* __restrict__ Wl,
    const float* __restrict__ bl, float* __restrict__ out) {
  __shared__ float s_pool[128];
  int c = threadIdx.x;
  int b = blockIdx.x;
  float mean = bn2sum[c] / CNT2;
  float var  = bn2sq[c] / CNT2 - mean * mean;
  float sc   = g2[c] * rsqrtf(var + EPS);
  float sh   = be2[c] - mean * sc;
  float accum = 0.f;
  const float* zb = z + (long)b * T_ * 128;
  for (int t = 0; t < T_; ++t) {
    float v = zb[t * 128 + c] * sc + sh;
    accum += v > 0.f ? v : 0.f;
  }
  s_pool[c] = accum * (1.0f / T_);
  __syncthreads();
  if (c < 10) {
    float o = bl[c];
#pragma unroll
    for (int k = 0; k < 128; ++k) o += s_pool[k] * Wl[k * 10 + c];
    out[b * 10 + c] = o;
  }
}

// ---------------------------------------------------------------------------

extern "C" void kernel_launch(void* const* d_in, const int* in_sizes, int n_in,
                              void* d_out, int out_size, void* d_ws,
                              size_t ws_size, hipStream_t stream) {
  (void)in_sizes; (void)n_in; (void)out_size; (void)ws_size;
  const float* x    = (const float*)d_in[0];
  const float* adj  = (const float*)d_in[1];
  const float* W0   = (const float*)d_in[2];
  const float* b0   = (const float*)d_in[3];
  const float* W1   = (const float*)d_in[4];
  const float* b1   = (const float*)d_in[5];
  const float* Wr   = (const float*)d_in[6];
  const float* br   = (const float*)d_in[7];
  const float* g_bn = (const float*)d_in[8];
  const float* be_bn= (const float*)d_in[9];
  const float* Watt = (const float*)d_in[10];
  const float* a_l  = (const float*)d_in[11];
  const float* a_r  = (const float*)d_in[12];
  const float* Wmh  = (const float*)d_in[13];
  const float* Wc   = (const float*)d_in[14];
  const float* g2   = (const float*)d_in[15];
  const float* be2  = (const float*)d_in[16];
  const float* Wl   = (const float*)d_in[17];
  const float* bl   = (const float*)d_in[18];
  float* out = (float*)d_out;

  char* ws = (char*)d_ws;
  _Float16* ypre = (_Float16*)(ws + 0);           // 32768*16*32 f16 = 32 MiB
  _Float16* hout = (_Float16*)(ws + 33554432);    // 32768*384 f16   = 24 MiB
  float*    z    = (float*)(ws + 58720256);       // 32768*128 f32   = 16 MiB
  float*    stats= (float*)(ws + 75497472);       // 384 f32
  float* bn1sum = stats, * bn1sq = stats + 32;
  float* bn2sum = stats + 64, * bn2sq = stats + 192;

  hipMemsetAsync(stats, 0, 384 * sizeof(float), stream);
  k_gcn0<<<NB / 4, 128, 0, stream>>>(x, adj, W0, b0, ypre, bn1sum, bn1sq);
  k_main<<<NB / 2, 64, 0, stream>>>(x, adj, W1, b1, Wr, br, g_bn, be_bn, Watt,
                                    a_l, a_r, Wmh, ypre, bn1sum, bn1sq, hout);
  k_conv<<<NB / 64, 128, 0, stream>>>(hout, Wc, z, bn2sum, bn2sq);
  k_head<<<B_, 128, 0, stream>>>(z, bn2sum, bn2sq, g2, be2, Wl, bl, out);
}
